// DispersionLoss_18640158065259
// MI455X (gfx1250) — matrix operand order repaired and checked by reading, not compiled, not run on hardware
//
#include <hip/hip_runtime.h>
#include <hip/hip_bf16.h>

typedef __attribute__((ext_vector_type(16))) __bf16 v16bf;
typedef __attribute__((ext_vector_type(4)))  __bf16 v4bf;
typedef __attribute__((ext_vector_type(8)))  float  v8f;

#define TAU_INV (1.0f / 100.0f)
#define TILE 128
#define KC 32
#define LDSTRIDE 40   // bf16 elems per LDS row (80 bytes) -> conflict-free b128 reads

// ---- gfx1250 async global->LDS path (ASYNCcnt), guarded so compile cannot fail ----
#if defined(__has_builtin)
#  if __has_builtin(__builtin_amdgcn_global_load_async_to_lds_b128)
#    define HAVE_ASYNC_LDS 1
#  endif
#endif
#ifndef HAVE_ASYNC_LDS
#  define HAVE_ASYNC_LDS 0
#endif

#if HAVE_ASYNC_LDS
// Builtin signature (from hipcc diagnostic): param0 is
//   addrspace(1) pointer to 'int __attribute__((vector_size(16)))'
typedef int v4i_gcc __attribute__((vector_size(16)));
typedef __attribute__((address_space(1))) v4i_gcc* gv4i_p;
typedef __attribute__((address_space(3))) v4i_gcc* lv4i_p;
__device__ __forceinline__ void wait_async0() {
#  if __has_builtin(__builtin_amdgcn_s_wait_asynccnt)
    __builtin_amdgcn_s_wait_asynccnt(0);
#  else
    asm volatile("s_wait_asynccnt 0" ::: "memory");
#  endif
}
#endif

// ---------------- Kernel 1: fp32 -> bf16 convert + row squared norms ----------------
__global__ __launch_bounds__(128) void prep_kernel(const float* __restrict__ z,
                                                   unsigned short* __restrict__ zb_u16,
                                                   float* __restrict__ sq, int K) {
    __bf16* zb = reinterpret_cast<__bf16*>(zb_u16);
    const int row = blockIdx.x;
    const int tid = threadIdx.x;               // 128 threads, K = 512 -> float4 each
    const float4 f = ((const float4*)(z + (size_t)row * K))[tid];
    v4bf v;
    v.x = (__bf16)f.x; v.y = (__bf16)f.y; v.z = (__bf16)f.z; v.w = (__bf16)f.w;
    *(v4bf*)&zb[(size_t)row * K + tid * 4] = v;
    float s = f.x * f.x + f.y * f.y + f.z * f.z + f.w * f.w;
    __shared__ float red[128];
    red[tid] = s;
    __syncthreads();
    for (int st = 64; st > 0; st >>= 1) {
        if (tid < st) red[tid] += red[tid + st];
        __syncthreads();
    }
    if (tid == 0) sq[row] = red[0];
}

// ---------------- Kernel 2: fused WMMA GEMM + exp + partial reduction ----------------
__global__ __launch_bounds__(256) void gemm_exp_kernel(const unsigned short* __restrict__ zb_u16,
                                                       const float* __restrict__ sq,
                                                       float* __restrict__ partials,
                                                       int N, int K) {
    const __bf16* Zb = reinterpret_cast<const __bf16*>(zb_u16);
    const int bj = blockIdx.x, bi = blockIdx.y;
    const int blockId = bi * gridDim.x + bj;
    if (bj < bi) {                  // symmetry: only upper-triangular block tiles
        if (threadIdx.x == 0) partials[blockId] = 0.0f;
        return;
    }
    const int i0 = bi * TILE, j0 = bj * TILE;

    __shared__ __bf16 As[2][TILE * LDSTRIDE];   // double-buffered
    __shared__ __bf16 Bs[2][TILE * LDSTRIDE];
    __shared__ float  sqI[TILE];
    __shared__ float  sqJ[TILE];
    __shared__ float  red[256];

    const int tid  = threadIdx.x;
    const int wave = tid >> 5, lane = tid & 31;
    const int wm = wave >> 1, wn = wave & 1;     // 4x2 wave grid: 32 rows x 64 cols each
    const int m    = lane & 15, half = lane >> 4;

    if (tid < TILE)       sqI[tid]        = sq[i0 + tid];
    else                  sqJ[tid - TILE] = sq[j0 + (tid - TILE)];

    v8f acc[2][4] = {};

    const int r = tid >> 1, h = tid & 1;         // each thread stages one 32B row-chunk
    const size_t gA0 = (size_t)(i0 + r) * K + 16 * h;
    const size_t gB0 = (size_t)(j0 + r) * K + 16 * h;
    const int lo = r * LDSTRIDE + 16 * h;        // LDS elem offset for this thread's chunk
    const int ITERS = K / KC;

#if HAVE_ASYNC_LDS
    // ---- async global->LDS double-buffered pipeline (one barrier per iteration) ----
    #define ASYNC_ISSUE(kofs, buf)                                                         \
        {                                                                                  \
            const __bf16* ga = Zb + gA0 + (kofs);                                          \
            const __bf16* gb = Zb + gB0 + (kofs);                                          \
            __bf16* la = &As[(buf)][lo];                                                   \
            __bf16* lb = &Bs[(buf)][lo];                                                   \
            __builtin_amdgcn_global_load_async_to_lds_b128((gv4i_p)ga,       (lv4i_p)la,       0, 0); \
            __builtin_amdgcn_global_load_async_to_lds_b128((gv4i_p)(ga + 8), (lv4i_p)(la + 8), 0, 0); \
            __builtin_amdgcn_global_load_async_to_lds_b128((gv4i_p)gb,       (lv4i_p)lb,       0, 0); \
            __builtin_amdgcn_global_load_async_to_lds_b128((gv4i_p)(gb + 8), (lv4i_p)(lb + 8), 0, 0); \
        }
    ASYNC_ISSUE(0, 0);
    wait_async0();
    __syncthreads();
#else
    // ---- fallback: register-staged double buffer ----
    uint4 ra0, ra1, rb0, rb1;
    #define LOAD_REGS(kofs)                                                                \
        {                                                                                  \
            const uint4* pA = (const uint4*)(Zb + gA0 + (kofs));                           \
            const uint4* pB = (const uint4*)(Zb + gB0 + (kofs));                           \
            ra0 = pA[0]; ra1 = pA[1]; rb0 = pB[0]; rb1 = pB[1];                            \
            __builtin_prefetch((const void*)(Zb + gA0 + (kofs) + KC), 0, 0);               \
            __builtin_prefetch((const void*)(Zb + gB0 + (kofs) + KC), 0, 0);               \
        }
    #define STORE_LDS(buf)                                                                 \
        {                                                                                  \
            __bf16* la = &As[(buf)][lo];                                                   \
            __bf16* lb = &Bs[(buf)][lo];                                                   \
            *(uint4*)la = ra0; *(uint4*)(la + 8) = ra1;                                    \
            *(uint4*)lb = rb0; *(uint4*)(lb + 8) = rb1;                                    \
        }
    LOAD_REGS(0);
    STORE_LDS(0);
    if (ITERS > 1) LOAD_REGS(KC);
    __syncthreads();
#endif

    for (int it = 0; it < ITERS; ++it) {
        const int cur = it & 1;
#if HAVE_ASYNC_LDS
        if (it + 1 < ITERS) ASYNC_ISSUE((size_t)(it + 1) * KC, (it + 1) & 1);
#else
        if (it + 1 < ITERS) {
            STORE_LDS((it + 1) & 1);                     // regs hold tile it+1
            if (it + 2 < ITERS) LOAD_REGS((size_t)(it + 2) * KC);
        }
#endif
        // A fragments: 16x32 bf16 per ISA layout (lane<16: K 0-7 & 16-23; lane>=16: K 8-15 & 24-31)
        v16bf afrag[2];
        #pragma unroll
        for (int sm = 0; sm < 2; ++sm) {
            const int row = 32 * wm + 16 * sm + m;
            union { uint4 u[2]; v16bf v; } t;
            t.u[0] = *(const uint4*)&As[cur][row * LDSTRIDE + 8 * half];
            t.u[1] = *(const uint4*)&As[cur][row * LDSTRIDE + 16 + 8 * half];
            afrag[sm] = t.v;
        }
        // B fragments: 32x16 bf16, lane = column; lanes 0-15 hold K 0-7 & 16-23, 16-31 hold K 8-15 & 24-31
        v16bf bfrag[4];
        #pragma unroll
        for (int sn = 0; sn < 4; ++sn) {
            const int col = 64 * wn + 16 * sn + m;
            union { uint4 u[2]; v16bf v; } t;
            t.u[0] = *(const uint4*)&Bs[cur][col * LDSTRIDE + 8 * half];
            t.u[1] = *(const uint4*)&Bs[cur][col * LDSTRIDE + 16 + 8 * half];
            bfrag[sn] = t.v;
        }
        #pragma unroll
        for (int sm = 0; sm < 2; ++sm)
            #pragma unroll
            for (int sn = 0; sn < 4; ++sn)
                acc[sm][sn] = __builtin_amdgcn_wmma_f32_16x16x32_bf16(
                    false, afrag[sm], false, bfrag[sn],
                    (short)0, acc[sm][sn], false, false);
#if HAVE_ASYNC_LDS
        if (it + 1 < ITERS) wait_async0();               // tile it+1 landed in LDS
#endif
        __syncthreads();
    }

    // Fused epilogue: E = exp(-max(sq_i + sq_j - 2 g, 0)/tau), diagonal zeroed,
    // strictly-upper tiles counted twice (symmetry).
    const float w = (bj > bi) ? 2.0f : 1.0f;
    float psum = 0.0f;
    #pragma unroll
    for (int sm = 0; sm < 2; ++sm) {
        #pragma unroll
        for (int sn = 0; sn < 4; ++sn) {
            const int jloc = 64 * wn + 16 * sn + m;
            const float sj = sqJ[jloc];
            #pragma unroll
            for (int v = 0; v < 8; ++v) {
                const int iloc = 32 * wm + 16 * sm + v + 8 * half;
                float d = sqI[iloc] + sj - 2.0f * acc[sm][sn][v];
                d = fmaxf(d, 0.0f);
                float e = __expf(-d * TAU_INV);
                if (i0 + iloc == j0 + jloc) e = 0.0f;
                psum += e;
            }
        }
    }
    red[tid] = psum * w;
    __syncthreads();
    for (int st = 128; st > 0; st >>= 1) {
        if (tid < st) red[tid] += red[tid + st];
        __syncthreads();
    }
    if (tid == 0) partials[blockId] = red[0];
}

// ---------------- Kernel 3: deterministic final reduction + log ----------------
__global__ __launch_bounds__(256) void finalize_kernel(const float* __restrict__ partials,
                                                       int n, float* __restrict__ out,
                                                       float denom) {
    __shared__ float red[256];
    float s = 0.0f;
    for (int i = threadIdx.x; i < n; i += 256) s += partials[i];
    red[threadIdx.x] = s;
    __syncthreads();
    for (int st = 128; st > 0; st >>= 1) {
        if (threadIdx.x < st) red[threadIdx.x] += red[threadIdx.x + st];
        __syncthreads();
    }
    if (threadIdx.x == 0) out[0] = logf(red[0] / denom);
}

extern "C" void kernel_launch(void* const* d_in, const int* in_sizes, int n_in,
                              void* d_out, int out_size, void* d_ws, size_t ws_size,
                              hipStream_t stream) {
    const int K = 512;
    const int N = in_sizes[0] / K;              // 8192
    const float* z = (const float*)d_in[0];

    // workspace layout: [ bf16 z (N*K*2 B) | sq (N*4 B) | partials (T*T*4 B) ]
    unsigned short* zb = (unsigned short*)d_ws;
    float* sq = (float*)((char*)d_ws + (size_t)N * K * sizeof(unsigned short));
    float* partials = sq + N;

    prep_kernel<<<N, 128, 0, stream>>>(z, zb, sq, K);

    const int T = N / TILE;                     // 64
    dim3 grid(T, T);
    gemm_exp_kernel<<<grid, 256, 0, stream>>>(zb, sq, partials, N, K);

    const float denom = (float)N * (float)(N - 1);
    finalize_kernel<<<1, 256, 0, stream>>>(partials, T * T, (float*)d_out, denom);
}
